// TransConv_22419729285549
// MI455X (gfx1250) — matrix operand order repaired and checked by reading, not compiled, not run on hardware
//
#include <hip/hip_runtime.h>
#include <cmath>

// ---------------------------------------------------------------------------
// Problem constants (match reference)
// ---------------------------------------------------------------------------
constexpr int NTOK = 4096;   // N
constexpr int INF  = 512;    // IN
constexpr int HD   = 256;    // HID
constexpr int NH   = 2;      // heads
constexpr int DKH  = 128;    // DK = HID/H
constexpr float LN_EPS = 1e-5f;
constexpr float INV_SQRT_DK = 0.08838834764831845f; // 1/sqrt(128)

typedef __attribute__((ext_vector_type(16))) __bf16 v16bf;
typedef __attribute__((ext_vector_type(8)))  float  v8f;

// fp32 -> bf16 round-to-nearest-even
__device__ __forceinline__ unsigned short f2bf(float f) {
  unsigned u = __float_as_uint(f);
  u += 0x7FFFu + ((u >> 16) & 1u);
  return (unsigned short)(u >> 16);
}

// Low 32 bits of a generic LDS pointer are the LDS byte offset (aperture rules)
__device__ __forceinline__ unsigned lds_off(const void* p) {
  return (unsigned)(uintptr_t)p;
}

// CDNA5 async global->LDS copy (16B per lane), tracked by ASYNCcnt.
__device__ __forceinline__ void async_copy_b128(const unsigned short* src, unsigned dst) {
  asm volatile("global_load_async_to_lds_b128 %0, %1, off"
               :: "v"(dst), "v"(src) : "memory");
}
// Wait until at most n async ops remain outstanding (in-order completion)
#define WAIT_ASYNC(n) asm volatile("s_wait_asynccnt " #n ::: "memory")

// -------------------------------------------------------------------------
// WMMA fragment loaders (wave32, 16-bit operands), per CDNA5 ISA layouts.
// A 16x32 (MxK): lanes 0-15 row=lane, elems0-7 K=kb..kb+7, elems8-15 K=kb+16..23
// B 32x16 (KxN): lanes 0-15 col=lane K=0..7,16..23 ; lanes 16-31 col=lane-16 K=8..15,24..31
// Tiles live in LDS; strides keep every 8-element chunk 16B aligned.
// -------------------------------------------------------------------------
__device__ __forceinline__ v16bf load_frag_a(const unsigned short* tile, int stride,
                                             int rowbase, int k0, int lane) {
  const int row = rowbase + (lane & 15);
  const unsigned short* p = tile + row * stride + k0 + ((lane >> 4) << 3);
  union { uint4 q[2]; v16bf v; } f;
  const uint4* p4 = (const uint4*)p;
  f.q[0] = p4[0];   // K +0..+7
  f.q[1] = p4[2];   // K +16..+23
  return f.v;
}

// B stored "transposed" in LDS: tileT[col][k]
__device__ __forceinline__ v16bf load_frag_b(const unsigned short* tileT, int stride,
                                             int colbase, int k0, int lane) {
  const int col = colbase + (lane & 15);
  const unsigned short* p = tileT + col * stride + k0 + ((lane >> 4) << 3);
  union { uint4 q[2]; v16bf v; } f;
  const uint4* p4 = (const uint4*)p;
  f.q[0] = p4[0];   // K +0..+7
  f.q[1] = p4[2];   // K +16..+23
  return f.v;
}

#define WMMA_BF16(A, B, C) \
  __builtin_amdgcn_wmma_f32_16x16x32_bf16(false, (A), false, (B), (short)0, (C), false, false)

// ---------------------------------------------------------------------------
// Prep: fp32 -> bf16 elementwise (vectorized x4)
// ---------------------------------------------------------------------------
__global__ __launch_bounds__(256) void cvt_bf16_kernel(
    const float* __restrict__ in, unsigned short* __restrict__ out, int n4) {
  const int i = blockIdx.x * 256 + threadIdx.x;
  if (i < n4) {
    const float4 v = ((const float4*)in)[i];
    union { unsigned short u[4]; uint2 q; } t;
    t.u[0] = f2bf(v.x); t.u[1] = f2bf(v.y); t.u[2] = f2bf(v.z); t.u[3] = f2bf(v.w);
    ((uint2*)out)[i] = t.q;
  }
}

// Prep: W[K][N] fp32 -> WT[N][K] bf16 (32x32 LDS tiles)
__global__ __launch_bounds__(256) void transpose_bf16_kernel(
    const float* __restrict__ W, unsigned short* __restrict__ WT, int K, int N) {
  __shared__ unsigned short t[32][33];
  const int k0 = blockIdx.x * 32, n0 = blockIdx.y * 32;
  const int tx = threadIdx.x & 31, ty = threadIdx.x >> 5;  // 8 rows / pass
  #pragma unroll
  for (int p = 0; p < 4; ++p)
    t[ty + p * 8][tx] = f2bf(W[(k0 + ty + p * 8) * N + n0 + tx]);
  __syncthreads();
  #pragma unroll
  for (int p = 0; p < 4; ++p)
    WT[(n0 + ty + p * 8) * K + k0 + tx] = t[tx][ty + p * 8];
}

// ---------------------------------------------------------------------------
// GEMM: C[M,N] = A[M,K](bf16) @ B (given as BT[N][K] bf16) + bias[N]
// OUT_MODE: 0 = fp32, 1 = bf16, 2 = bf16 transposed C[n][m]
// block = 256 threads = 8 waves, tile 64x64, K-step 32.
// Tiles streamed with async-to-LDS, double-buffered (2 async/thread/step).
// ---------------------------------------------------------------------------
template <int OUT_MODE>
__global__ __launch_bounds__(256) void gemm_kernel(
    const unsigned short* __restrict__ A, const unsigned short* __restrict__ BT,
    const float* __restrict__ bias, void* __restrict__ Cp,
    int M, int K, int N) {
  __shared__ __align__(16) unsigned short As[2][64 * 40];  // [m][k]
  __shared__ __align__(16) unsigned short Bs[2][64 * 40];  // [n][k]

  const int tid = threadIdx.x, lane = tid & 31, wave = tid >> 5;
  const int m0 = blockIdx.x * 64, n0 = blockIdx.y * 64;
  const int wr = (wave >> 2) * 32;   // 0 or 32 (two 16-row tiles)
  const int wc = (wave & 3) * 16;    // 16-col tile

  const int r = tid >> 2, c8 = (tid & 3) << 3;   // one 16B chunk each
  auto issue = [&](int kk, int buf) {
    async_copy_b128(A  + (m0 + r) * K + kk + c8, lds_off(&As[buf][r * 40 + c8]));
    async_copy_b128(BT + (n0 + r) * K + kk + c8, lds_off(&Bs[buf][r * 40 + c8]));
  };

  v8f acc0 = {}; v8f acc1 = {};
  const int nk = K >> 5;
  issue(0, 0);
  for (int i = 0; i < nk; ++i) {
    const int cur = i & 1;
    issue(((i + 1) == nk ? 0 : (i + 1) << 5), cur ^ 1);  // prefetch (wraps)
    WAIT_ASYNC(2);
    __syncthreads();
    v16bf bf = load_frag_b(Bs[cur], 40, wc, 0, lane);
    v16bf a0 = load_frag_a(As[cur], 40, wr, 0, lane);
    v16bf a1 = load_frag_a(As[cur], 40, wr + 16, 0, lane);
    acc0 = WMMA_BF16(a0, bf, acc0);
    acc1 = WMMA_BF16(a1, bf, acc1);
    __syncthreads();
  }

  // C layout: VGPR r -> row (lane<16 ? r : 8+r), col = lane&15
  const int colg = n0 + wc + (lane & 15);
  const float bv = bias[colg];
  const int rbase = (lane >> 4) << 3;

  if constexpr (OUT_MODE == 2) {
    // C[n][m] bf16: each acc's 8 rows are contiguous -> single 16B store
    unsigned short* Cb = (unsigned short*)Cp;
    union { unsigned short u[8]; uint4 q; } t0, t1;
    #pragma unroll
    for (int q = 0; q < 8; ++q) {
      t0.u[q] = f2bf(acc0[q] + bv);
      t1.u[q] = f2bf(acc1[q] + bv);
    }
    const int base = colg * M + m0 + wr + rbase;
    *(uint4*)&Cb[base]      = t0.q;
    *(uint4*)&Cb[base + 16] = t1.q;
  } else if constexpr (OUT_MODE == 1) {
    unsigned short* Cb = (unsigned short*)Cp;
    #pragma unroll
    for (int q = 0; q < 8; ++q) {
      Cb[(m0 + wr + rbase + q) * N + colg]      = f2bf(acc0[q] + bv);
      Cb[(m0 + wr + 16 + rbase + q) * N + colg] = f2bf(acc1[q] + bv);
    }
  } else {
    float* C = (float*)Cp;
    #pragma unroll
    for (int q = 0; q < 8; ++q) {
      C[(m0 + wr + rbase + q) * N + colg]      = acc0[q] + bv;
      C[(m0 + wr + 16 + rbase + q) * N + colg] = acc1[q] + bv;
    }
  }
}

// ---------------------------------------------------------------------------
// Block-wide sum over 256 threads: wave32 shuffle reduce + 8-partial LDS pass
// ---------------------------------------------------------------------------
__device__ __forceinline__ float block_sum256(float v, float* red) {
  const int tid = threadIdx.x;
  #pragma unroll
  for (int m = 16; m >= 1; m >>= 1) v += __shfl_xor(v, m, 32);
  __syncthreads();
  if ((tid & 31) == 0) red[tid >> 5] = v;
  __syncthreads();
  float s = 0.f;
  #pragma unroll
  for (int i = 0; i < 8; ++i) s += red[i];
  return s;
}

// Row LayerNorm (+optional ReLU); writes fp32 + bf16 shadow
__global__ __launch_bounds__(256) void ln_kernel(
    const float* __restrict__ in, const float* __restrict__ g,
    const float* __restrict__ b, float* __restrict__ out,
    unsigned short* __restrict__ out16, int relu) {
  __shared__ float red[8];
  const int n = blockIdx.x, d = threadIdx.x;
  const float v = in[n * HD + d];
  const float mean = block_sum256(v, red) * (1.0f / HD);
  const float c = v - mean;
  const float var = block_sum256(c * c, red) * (1.0f / HD);
  float r = c * rsqrtf(var + LN_EPS) * g[d] + b[d];
  if (relu) r = fmaxf(r, 0.0f);
  out[n * HD + d] = r;
  out16[n * HD + d] = f2bf(r);
}

// x' = LN( 0.5 * mean_h(out[n,h,:]) + 0.5 * x ); writes fp32 + bf16 shadow
__global__ __launch_bounds__(256) void blend_ln_kernel(
    const float* __restrict__ ob, const float* __restrict__ xin,
    const float* __restrict__ g, const float* __restrict__ b,
    float* __restrict__ out, unsigned short* __restrict__ out16) {
  __shared__ float red[8];
  const int n = blockIdx.x, d = threadIdx.x;
  const float om = 0.5f * (ob[n * (NH * HD) + d] + ob[n * (NH * HD) + HD + d]);
  const float v = 0.5f * om + 0.5f * xin[n * HD + d];
  const float mean = block_sum256(v, red) * (1.0f / HD);
  const float c = v - mean;
  const float var = block_sum256(c * c, red) * (1.0f / HD);
  const float r = c * rsqrtf(var + LN_EPS) * g[d] + b[d];
  out[n * HD + d] = r;
  out16[n * HD + d] = f2bf(r);
}

// ---------------------------------------------------------------------------
// Fused attention (head-softmax is elementwise in (n,l)):
//   s[n,l,h'] = (q[n,h',:].k[l,h',:]) / sqrt(dk)
//   w[n,l,h]  = exp(s_h) / (exp(s_0)+exp(s_1))
//   O[n,h,:] += w[n,l,h] * v[l,h,:]
// Inputs: qb/kb bf16 [N][256]; vbT bf16 transposed [h*HD+d][N].
// Block: 32 output rows x 1 head, 256 threads = 8 waves.
// K/V tiles double-buffered via async-to-LDS (8 async/thread/tile).
// ---------------------------------------------------------------------------
__global__ __launch_bounds__(256) void attn_kernel(
    const unsigned short* __restrict__ qb, const unsigned short* __restrict__ kb,
    const unsigned short* __restrict__ vbT, float* __restrict__ ob) {
  const int n0 = blockIdx.x * 32;
  const int h  = blockIdx.y;
  const int tid = threadIdx.x, lane = tid & 31, wave = tid >> 5;

  __shared__ __align__(16) unsigned short Qs[32 * 264];      // [row][feat] both heads
  __shared__ __align__(16) unsigned short Ks[2][32 * 264];   // [l][feat] both heads
  __shared__ __align__(16) unsigned short Vt[2][256 * 40];   // [d][l] (head h)
  __shared__ __align__(16) unsigned short Ws[32 * 40];       // bf16 weights head h
  __shared__ float Sld[2][32][33];                           // raw scores

  auto issue_kv = [&](int l0, int buf) {
    #pragma unroll
    for (int p = 0; p < 4; ++p) {      // K tile: 32 x 256 bf16
      const int chunk = p * 256 + tid;
      const int r = chunk >> 5, c8 = (chunk & 31) << 3;
      async_copy_b128(kb + (l0 + r) * HD + c8, lds_off(&Ks[buf][r * 264 + c8]));
    }
    #pragma unroll
    for (int p = 0; p < 4; ++p) {      // V tile: 256 d x 32 l bf16 (pre-transposed)
      const int chunk = p * 256 + tid;
      const int d = chunk >> 2, c8 = (chunk & 3) << 3;
      async_copy_b128(vbT + (h * HD + d) * NTOK + l0 + c8,
                      lds_off(&Vt[buf][d * 40 + c8]));
    }
  };

  // stage Q tile (32 x 256 bf16) + first K/V tile
  #pragma unroll
  for (int p = 0; p < 4; ++p) {
    const int chunk = p * 256 + tid;
    const int r = chunk >> 5, c8 = (chunk & 31) << 3;
    async_copy_b128(qb + (n0 + r) * HD + c8, lds_off(&Qs[r * 264 + c8]));
  }
  issue_kv(0, 0);

  // score-wave roles: 2 row tiles x 2 l tiles x 2 heads
  const int srt = (wave & 1) * 16;
  const int slt = ((wave >> 1) & 1) * 16;
  const int sh  = wave >> 2;
  // output-wave roles: 2 row tiles x 4 col groups (64 cols each)
  const int orow = (wave & 1) * 16;
  const int ocol = (wave >> 1) * 64;

  v8f acc[4] = {};

  for (int l0 = 0; l0 < NTOK; l0 += 32) {
    const int cur = (l0 >> 5) & 1;
    issue_kv((l0 + 32) & (NTOK - 1), cur ^ 1);  // prefetch next tile (wraps)
    WAIT_ASYNC(8);          // everything older than the prefetch is complete
    __syncthreads();

    // scores for head sh: S = Q_sh * K_sh^T  (K=128 -> 4 WMMA steps)
    v8f s = {};
    #pragma unroll
    for (int ks = 0; ks < 4; ++ks) {
      v16bf a  = load_frag_a(Qs, 264, srt, sh * DKH + ks * 32, lane);
      v16bf bf = load_frag_b(Ks[cur], 264, slt, sh * DKH + ks * 32, lane);
      s = WMMA_BF16(a, bf, s);
    }
    { // spill to LDS with 1/sqrt(dk) scale
      const int rbase = (lane >> 4) << 3, cc = lane & 15;
      #pragma unroll
      for (int q = 0; q < 8; ++q)
        Sld[sh][srt + rbase + q][slt + cc] = s[q] * INV_SQRT_DK;
    }
    __syncthreads();

    // elementwise 2-way head softmax -> bf16 weight tile for head h
    #pragma unroll
    for (int e = 0; e < 4; ++e) {
      const int idx = e * 256 + tid;
      const int rr = idx >> 5, c = idx & 31;
      const float s0 = Sld[0][rr][c], s1 = Sld[1][rr][c];
      const float m = fmaxf(s0, s1);
      const float e0 = __expf(s0 - m), e1 = __expf(s1 - m);
      const float wgt = (h == 0 ? e0 : e1) / (e0 + e1);
      Ws[rr * 40 + c] = f2bf(wgt);
    }
    __syncthreads();

    // O += W * V   (K = 32 over the l-tile)
    v16bf aw = load_frag_a(Ws, 40, orow, 0, lane);
    #pragma unroll
    for (int t = 0; t < 4; ++t) {
      v16bf bv = load_frag_b(Vt[cur], 40, ocol + t * 16, 0, lane);
      acc[t] = WMMA_BF16(aw, bv, acc[t]);
    }
    __syncthreads();        // done with cur before next iteration reuses it
  }

  // store O tile for this head (fp32)
  const int rbase = (lane >> 4) << 3, cc = lane & 15;
  #pragma unroll
  for (int t = 0; t < 4; ++t)
    #pragma unroll
    for (int q = 0; q < 8; ++q)
      ob[(n0 + orow + rbase + q) * (NH * HD) + h * HD + ocol + t * 16 + cc] =
          acc[t][q];
}

// ---------------------------------------------------------------------------
// Orchestration
// ---------------------------------------------------------------------------
extern "C" void kernel_launch(void* const* d_in, const int* in_sizes, int n_in,
                              void* d_out, int out_size, void* d_ws, size_t ws_size,
                              hipStream_t stream) {
  const float* x    = (const float*)d_in[0];   // [4096,512]
  const float* fcw  = (const float*)d_in[1];   // [512,256]
  const float* fcb  = (const float*)d_in[2];   // [256]
  const float* lng  = (const float*)d_in[3];   // [3,256]
  const float* lnb  = (const float*)d_in[4];   // [3,256]
  const float* wq   = (const float*)d_in[5];   // [2,256,256]
  const float* bq   = (const float*)d_in[6];   // [2,256]
  const float* wk   = (const float*)d_in[7];
  const float* bk   = (const float*)d_in[8];
  const float* wvl  = (const float*)d_in[9];
  const float* bvl  = (const float*)d_in[10];
  const float* wvg  = (const float*)d_in[11];  // [2,256,512]
  const float* bvg  = (const float*)d_in[12];  // [2,512]

  char* w = (char*)d_ws;
  const size_t MB = 1u << 20;
  float* xA            = (float*)(w);             // 4 MB
  float* xB            = (float*)(w + 4 * MB);    // 4 MB
  float* yln           = (float*)(w + 8 * MB);    // 4 MB  fc pre-LN
  float* ob            = (float*)(w + 12 * MB);   // 8 MB  attn out [N,2*HD]
  unsigned short* qb16 = (unsigned short*)(w + 20 * MB);  // 2 MB
  unsigned short* kb16 = (unsigned short*)(w + 22 * MB);  // 2 MB
  unsigned short* vi16 = (unsigned short*)(w + 24 * MB);  // 2 MB
  unsigned short* vbT  = (unsigned short*)(w + 26 * MB);  // 4 MB [2*HD][N]
  unsigned short* x16  = (unsigned short*)(w + 30 * MB);  // 4 MB
  unsigned short* xA16 = (unsigned short*)(w + 34 * MB);  // 2 MB
  unsigned short* xB16 = (unsigned short*)(w + 36 * MB);  // 2 MB
  unsigned short* fcwT = (unsigned short*)(w + 38 * MB);  // [256][512]
  unsigned short* wqT  = fcwT + INF * HD;                 // 2 x [256][256]
  unsigned short* wkT  = wqT + 2 * HD * HD;
  unsigned short* wvlT = wkT + 2 * HD * HD;
  unsigned short* wvgT = wvlT + 2 * HD * HD;              // 2 x [512][256]

  const dim3 blk(256);

  // ---- one-time prep: bf16 conversions / weight transposes ----
  cvt_bf16_kernel<<<(NTOK * INF / 4 + 255) / 256, blk, 0, stream>>>(x, x16, NTOK * INF / 4);
  transpose_bf16_kernel<<<dim3(INF / 32, HD / 32), blk, 0, stream>>>(fcw, fcwT, INF, HD);
  for (int l = 0; l < 2; ++l) {
    transpose_bf16_kernel<<<dim3(HD / 32, HD / 32), blk, 0, stream>>>(
        wq + l * HD * HD, wqT + l * HD * HD, HD, HD);
    transpose_bf16_kernel<<<dim3(HD / 32, HD / 32), blk, 0, stream>>>(
        wk + l * HD * HD, wkT + l * HD * HD, HD, HD);
    transpose_bf16_kernel<<<dim3(HD / 32, HD / 32), blk, 0, stream>>>(
        wvl + l * HD * HD, wvlT + l * HD * HD, HD, HD);
    transpose_bf16_kernel<<<dim3(HD / 32, (NH * HD) / 32), blk, 0, stream>>>(
        wvg + l * HD * NH * HD, wvgT + l * HD * NH * HD, HD, NH * HD);
  }

  // ---- fc + LN + ReLU ----
  gemm_kernel<0><<<dim3(NTOK / 64, HD / 64), blk, 0, stream>>>(
      x16, fcwT, fcb, yln, NTOK, INF, HD);
  ln_kernel<<<NTOK, blk, 0, stream>>>(yln, lng, lnb, xA, xA16, 1);

  const float* xin = xA;
  const unsigned short* xin16 = xA16;
  for (int l = 0; l < 2; ++l) {
    const int wo = l * HD * HD;
    gemm_kernel<1><<<dim3(NTOK / 64, HD / 64), blk, 0, stream>>>(
        xin16, wqT + wo, bq + l * HD, qb16, NTOK, HD, HD);
    gemm_kernel<1><<<dim3(NTOK / 64, HD / 64), blk, 0, stream>>>(
        xin16, wkT + wo, bk + l * HD, kb16, NTOK, HD, HD);
    gemm_kernel<1><<<dim3(NTOK / 64, HD / 64), blk, 0, stream>>>(
        xin16, wvlT + wo, bvl + l * HD, vi16, NTOK, HD, HD);
    // v = vin @ wvg + bvg, written bf16 *transposed* [2*HD][N]
    gemm_kernel<2><<<dim3(NTOK / 64, (NH * HD) / 64), blk, 0, stream>>>(
        vi16, wvgT + l * HD * NH * HD, bvg + l * NH * HD, vbT, NTOK, HD, NH * HD);

    attn_kernel<<<dim3(NTOK / 32, NH), blk, 0, stream>>>(qb16, kb16, vbT, ob);

    float* xout = (l == 0) ? xB : (float*)d_out;
    unsigned short* xout16 = (l == 0) ? xB16 : xA16;  // last-layer shadow unused
    blend_ln_kernel<<<NTOK, blk, 0, stream>>>(
        ob, xin, lng + (l + 1) * HD, lnb + (l + 1) * HD, xout, xout16);
    xin = xout;
    xin16 = xout16;
  }
}
